// TaskAllocatorNetwork_35983236006575
// MI455X (gfx1250) — matrix operand order repaired and checked by reading, not compiled, not run on hardware
//
#include <hip/hip_runtime.h>

typedef __attribute__((ext_vector_type(16))) _Float16 v16h;
typedef __attribute__((ext_vector_type(8)))  _Float16 v8h;
typedef __attribute__((ext_vector_type(8)))  float    v8f;

#define B_  32
#define R_  512
#define T_  128
#define H_  64

// single-instruction relu: v_med3_f32(x, 0, BIG)
__device__ __forceinline__ float relu1(float x) {
    return __builtin_amdgcn_fmed3f(x, 0.f, 3.0e38f);
}

// ---------------------------------------------------------------------------
// K1: robot encoder + fused wk/wv/a_w1_hi projections.
// rf = relu(relu(x@W1+b1)@W2+b2);  kmat=rf@wk+bk; vmat=rf@wv+bv;
// rp  = (f16) rf@a_w1[H:]
// 64 rows per block, 256 threads (4 threads/row, 16 neurons each).
// ---------------------------------------------------------------------------
__global__ __launch_bounds__(256) void robot_encoder_kernel(
    const float* __restrict__ x_in,
    const float* __restrict__ w1, const float* __restrict__ b1,
    const float* __restrict__ w2, const float* __restrict__ b2,
    const float* __restrict__ wk, const float* __restrict__ bk,
    const float* __restrict__ wv, const float* __restrict__ bv,
    const float* __restrict__ a_w1,
    float* __restrict__ kmat, float* __restrict__ vmat,
    _Float16* __restrict__ rp)
{
    __shared__ float s_w1[7][64];
    __shared__ float s_w2[64][64];
    __shared__ float s_wk[64][64];
    __shared__ float s_wv[64][64];
    __shared__ float s_wa[64][64];
    __shared__ float s_b1[64], s_b2[64], s_bk[64], s_bv[64];
    __shared__ float s_x[64][8];
    __shared__ float s_h[64][64];
    __shared__ float s_rf[64][64];

    const int tid = threadIdx.x;
    const int b   = blockIdx.x >> 3;
    const int r0  = (blockIdx.x & 7) * 64;

    for (int i = tid; i < 7 * 64; i += 256) s_w1[i / 64][i % 64] = w1[i];
    for (int i = tid; i < 4096; i += 256) {
        const int k = i >> 6, n = i & 63;
        s_w2[k][n] = w2[i];
        s_wk[k][n] = wk[i];
        s_wv[k][n] = wv[i];
        s_wa[k][n] = a_w1[4096 + i];   // upper half rows [H:2H)
    }
    if (tid < 64) { s_b1[tid] = b1[tid]; s_b2[tid] = b2[tid]; s_bk[tid] = bk[tid]; s_bv[tid] = bv[tid]; }
    for (int i = tid; i < 64 * 7; i += 256) {
        const int rr = i / 7, k = i % 7;
        s_x[rr][k] = x_in[((long)(b * R_) + r0 + rr) * 7 + k];
    }
    __syncthreads();

    const int row = tid >> 2;
    const int n0  = (tid & 3) << 4;
    float acc[16];

    // layer 1 (K=7)
    #pragma unroll
    for (int j = 0; j < 16; ++j) {
        float a = s_b1[n0 + j];
        #pragma unroll
        for (int k = 0; k < 7; ++k) a += s_x[row][k] * s_w1[k][n0 + j];
        s_h[row][n0 + j] = relu1(a);
    }
    __syncthreads();

    // layer 2 (K=64)
    #pragma unroll
    for (int j = 0; j < 16; ++j) acc[j] = s_b2[n0 + j];
    for (int k = 0; k < 64; ++k) {
        const float xv = s_h[row][k];
        #pragma unroll
        for (int j = 0; j < 16; ++j) acc[j] += xv * s_w2[k][n0 + j];
    }
    #pragma unroll
    for (int j = 0; j < 16; ++j) s_rf[row][n0 + j] = relu1(acc[j]);
    __syncthreads();

    const long base = ((long)(b * R_) + r0 + row) * H_ + n0;

    // wk projection
    #pragma unroll
    for (int j = 0; j < 16; ++j) acc[j] = s_bk[n0 + j];
    for (int k = 0; k < 64; ++k) {
        const float xv = s_rf[row][k];
        #pragma unroll
        for (int j = 0; j < 16; ++j) acc[j] += xv * s_wk[k][n0 + j];
    }
    #pragma unroll
    for (int j = 0; j < 16; ++j) kmat[base + j] = acc[j];

    // wv projection
    #pragma unroll
    for (int j = 0; j < 16; ++j) acc[j] = s_bv[n0 + j];
    for (int k = 0; k < 64; ++k) {
        const float xv = s_rf[row][k];
        #pragma unroll
        for (int j = 0; j < 16; ++j) acc[j] += xv * s_wv[k][n0 + j];
    }
    #pragma unroll
    for (int j = 0; j < 16; ++j) vmat[base + j] = acc[j];

    // rp = rf @ a_w1[H:] (no bias), stored f16 for the WMMA pairwise stage
    #pragma unroll
    for (int j = 0; j < 16; ++j) acc[j] = 0.f;
    for (int k = 0; k < 64; ++k) {
        const float xv = s_rf[row][k];
        #pragma unroll
        for (int j = 0; j < 16; ++j) acc[j] += xv * s_wa[k][n0 + j];
    }
    v16h hv;
    #pragma unroll
    for (int j = 0; j < 16; ++j) hv[j] = (_Float16)acc[j];
    *(v16h*)(rp + base) = hv;
}

// ---------------------------------------------------------------------------
// K2: task encoder + fused wq projection.  q = relu(relu(x@W1)@W2) @ wq + bq
// ---------------------------------------------------------------------------
__global__ __launch_bounds__(256) void task_encoder_kernel(
    const float* __restrict__ x_in,
    const float* __restrict__ w1, const float* __restrict__ b1,
    const float* __restrict__ w2, const float* __restrict__ b2,
    const float* __restrict__ wq, const float* __restrict__ bq,
    float* __restrict__ qbuf)
{
    __shared__ float s_w1[6][64];
    __shared__ float s_w2[64][64];
    __shared__ float s_wq[64][64];
    __shared__ float s_b1[64], s_b2[64], s_bq[64];
    __shared__ float s_x[64][8];
    __shared__ float s_h[64][64];
    __shared__ float s_tf[64][64];

    const int tid = threadIdx.x;
    const int b   = blockIdx.x >> 1;
    const int t0  = (blockIdx.x & 1) * 64;

    for (int i = tid; i < 6 * 64; i += 256) s_w1[i / 64][i % 64] = w1[i];
    for (int i = tid; i < 4096; i += 256) {
        const int k = i >> 6, n = i & 63;
        s_w2[k][n] = w2[i];
        s_wq[k][n] = wq[i];
    }
    if (tid < 64) { s_b1[tid] = b1[tid]; s_b2[tid] = b2[tid]; s_bq[tid] = bq[tid]; }
    for (int i = tid; i < 64 * 6; i += 256) {
        const int rr = i / 6, k = i % 6;
        s_x[rr][k] = x_in[((long)(b * T_) + t0 + rr) * 6 + k];
    }
    __syncthreads();

    const int row = tid >> 2;
    const int n0  = (tid & 3) << 4;
    float acc[16];

    #pragma unroll
    for (int j = 0; j < 16; ++j) {
        float a = s_b1[n0 + j];
        #pragma unroll
        for (int k = 0; k < 6; ++k) a += s_x[row][k] * s_w1[k][n0 + j];
        s_h[row][n0 + j] = relu1(a);
    }
    __syncthreads();

    #pragma unroll
    for (int j = 0; j < 16; ++j) acc[j] = s_b2[n0 + j];
    for (int k = 0; k < 64; ++k) {
        const float xv = s_h[row][k];
        #pragma unroll
        for (int j = 0; j < 16; ++j) acc[j] += xv * s_w2[k][n0 + j];
    }
    #pragma unroll
    for (int j = 0; j < 16; ++j) s_tf[row][n0 + j] = relu1(acc[j]);
    __syncthreads();

    #pragma unroll
    for (int j = 0; j < 16; ++j) acc[j] = s_bq[n0 + j];
    for (int k = 0; k < 64; ++k) {
        const float xv = s_tf[row][k];
        #pragma unroll
        for (int j = 0; j < 16; ++j) acc[j] += xv * s_wq[k][n0 + j];
    }
    const long base = ((long)(b * T_) + t0 + row) * H_ + n0;
    #pragma unroll
    for (int j = 0; j < 16; ++j) qbuf[base + j] = acc[j];
}

// ---------------------------------------------------------------------------
// K3: attention, one block per (batch, head), one thread per task (128),
// online (flash-style) softmax over the 512 robots. k/v staged in LDS.
// ---------------------------------------------------------------------------
__global__ __launch_bounds__(128) void attn_kernel(
    const float* __restrict__ qbuf,
    const float* __restrict__ kmat,
    const float* __restrict__ vmat,
    float* __restrict__ ctx)
{
    __shared__ float sk[R_][16];
    __shared__ float sv[R_][16];
    const int b = blockIdx.x >> 2;
    const int h = blockIdx.x & 3;
    const int tid = threadIdx.x;

    for (int i = tid; i < R_ * 4; i += 128) {     // 2048 float4s
        const int r = i >> 2, c4 = (i & 3) * 4;
        const long g = ((long)(b * R_) + r) * H_ + h * 16 + c4;
        *(float4*)(&sk[r][c4]) = *(const float4*)(kmat + g);
        *(float4*)(&sv[r][c4]) = *(const float4*)(vmat + g);
    }
    __syncthreads();

    float qv[16];
    const float* qrow = qbuf + ((long)(b * T_) + tid) * H_ + h * 16;
    #pragma unroll
    for (int d = 0; d < 16; ++d) qv[d] = qrow[d] * 0.25f;   // 1/sqrt(16)

    float m = -3.0e38f, s = 0.f;
    float acc[16];
    #pragma unroll
    for (int d = 0; d < 16; ++d) acc[d] = 0.f;

    for (int r = 0; r < R_; ++r) {
        float logit = 0.f;
        #pragma unroll
        for (int d = 0; d < 16; ++d) logit += qv[d] * sk[r][d];
        const float nm   = fmaxf(m, logit);
        const float corr = __expf(m - nm);
        const float p    = __expf(logit - nm);
        s = s * corr + p;
        #pragma unroll
        for (int d = 0; d < 16; ++d) acc[d] = acc[d] * corr + p * sv[r][d];
        m = nm;
    }
    const float inv = 1.f / s;
    float* crow = ctx + ((long)(b * T_) + tid) * H_ + h * 16;
    #pragma unroll
    for (int d = 0; d < 16; ++d) crow[d] = acc[d] * inv;
}

// ---------------------------------------------------------------------------
// K4: tf = ctx@wo + bo (no relu); tp = (f16)(tf@a_w1[:H] + a_b1)
// ---------------------------------------------------------------------------
__global__ __launch_bounds__(256) void outproj_tp_kernel(
    const float* __restrict__ ctx,
    const float* __restrict__ wo, const float* __restrict__ bo,
    const float* __restrict__ a_w1, const float* __restrict__ a_b1,
    _Float16* __restrict__ tp)
{
    __shared__ float s_wo[64][64];
    __shared__ float s_wa[64][64];
    __shared__ float s_bo[64], s_ab1[64];
    __shared__ float s_ctx[64][64];
    __shared__ float s_tf[64][64];

    const int tid = threadIdx.x;
    const int b   = blockIdx.x >> 1;
    const int t0  = (blockIdx.x & 1) * 64;

    for (int i = tid; i < 4096; i += 256) {
        const int k = i >> 6, n = i & 63;
        s_wo[k][n] = wo[i];
        s_wa[k][n] = a_w1[i];                     // lower half rows [0:H)
    }
    if (tid < 64) { s_bo[tid] = bo[tid]; s_ab1[tid] = a_b1[tid]; }
    const long cbase = ((long)(b * T_) + t0) * H_;
    for (int i = tid; i < 4096; i += 256) s_ctx[i >> 6][i & 63] = ctx[cbase + i];
    __syncthreads();

    const int row = tid >> 2;
    const int n0  = (tid & 3) << 4;
    float acc[16];

    #pragma unroll
    for (int j = 0; j < 16; ++j) acc[j] = s_bo[n0 + j];
    for (int k = 0; k < 64; ++k) {
        const float xv = s_ctx[row][k];
        #pragma unroll
        for (int j = 0; j < 16; ++j) acc[j] += xv * s_wo[k][n0 + j];
    }
    #pragma unroll
    for (int j = 0; j < 16; ++j) s_tf[row][n0 + j] = acc[j];   // no relu
    __syncthreads();

    #pragma unroll
    for (int j = 0; j < 16; ++j) acc[j] = s_ab1[n0 + j];
    for (int k = 0; k < 64; ++k) {
        const float xv = s_tf[row][k];
        #pragma unroll
        for (int j = 0; j < 16; ++j) acc[j] += xv * s_wa[k][n0 + j];
    }
    v16h hv;
    #pragma unroll
    for (int j = 0; j < 16; ++j) hv[j] = (_Float16)acc[j];
    *(v16h*)(tp + cbase + row * 64 + n0) = hv;
}

// ---------------------------------------------------------------------------
// K5: pairwise MLP via WMMA + row softmax.
// Block = (batch, 16-task tile). 8 waves; wave w owns robots [w*64, w*64+64),
// processed 2 per iteration. tp/rp are pre-converted f16, so the A build is
// pure packed-f16 add+max (no cvt). a_b2 is folded into the WMMA C operand.
// relu(D).a_w3 (relu as single v_med3_f32) is reduced with a reduce-scatter
// butterfly (15 permutes per 2 robots, every lane ends with exactly one
// score) -> padded LDS -> row softmax.
// ---------------------------------------------------------------------------
#define SCP 516   // row pad: (4m + r) mod 64 distinct over all 32 store lanes

__global__ __launch_bounds__(256) void pairwise_kernel(
    const _Float16* __restrict__ tp,  // [B,T,64] f16, a_b1 folded in
    const _Float16* __restrict__ rp,  // [B,R,64] f16
    const float* __restrict__ a_w2,   // [64,32]
    const float* __restrict__ a_b2,   // [32]
    const float* __restrict__ a_w3,   // [32]
    float* __restrict__ out)          // [B,T,R]
{
    __shared__ float sc[16][SCP];

    const int b     = blockIdx.x >> 3;
    const int t0    = (blockIdx.x & 7) << 4;
    const int tid   = threadIdx.x;
    const int lane  = tid & 31;
    const int wid   = tid >> 5;     // 0..7
    const int lhalf = lane >> 4;    // 0/1
    const int l16   = lane & 15;

    // B operands: Bop[kh][nh], ISA B layout: k = 32*kh + 16*(e>>3) + 8*lhalf + (e&7), n = 16*nh + l16
    v16h Bop[2][2];
    #pragma unroll
    for (int kh = 0; kh < 2; ++kh)
        #pragma unroll
        for (int nh = 0; nh < 2; ++nh)
            #pragma unroll
            for (int e = 0; e < 16; ++e)
                Bop[kh][nh][e] =
                    (_Float16)a_w2[(32 * kh + 16 * (e >> 3) + 8 * lhalf + (e & 7)) * 32 + 16 * nh + l16];

    // tp in A-operand pattern: for K-half kh: halves at k = 32*kh + 16*c + 8*lhalf + u
    v16h tph[2];
    {
        const _Float16* trow = tp + ((long)(b * T_) + t0 + l16) * H_;
        #pragma unroll
        for (int kh = 0; kh < 2; ++kh) {
            const v8h lo = *(const v8h*)(trow + 32 * kh + 8 * lhalf);
            const v8h hi = *(const v8h*)(trow + 32 * kh + 16 + 8 * lhalf);
            tph[kh] = __builtin_shufflevector(lo, hi,
                0, 1, 2, 3, 4, 5, 6, 7, 8, 9, 10, 11, 12, 13, 14, 15);
        }
    }
    const float w3lo = a_w3[l16], w3hi = a_w3[16 + l16];

    // a_b2 folded into the accumulator init (C layout: col n = l16 per lane)
    v8f C0, C1;
    {
        const float b2lo = a_b2[l16], b2hi = a_b2[16 + l16];
        #pragma unroll
        for (int p = 0; p < 8; ++p) { C0[p] = b2lo; C1[p] = b2hi; }
    }
    const v16h zf = {};

    const int rend = wid * 64 + 64;
    for (int r0 = wid * 64; r0 < rend; r0 += 2) {   // wave-uniform: EXEC all ones
        const _Float16* ra = rp + ((long)(b * R_) + r0) * H_;
        const _Float16* rb = ra + H_;
        v16h rpa[2], rpb[2];
        #pragma unroll
        for (int kh = 0; kh < 2; ++kh) {
            const v8h alo = *(const v8h*)(ra + 32 * kh + 8 * lhalf);
            const v8h ahi = *(const v8h*)(ra + 32 * kh + 16 + 8 * lhalf);
            const v8h blo = *(const v8h*)(rb + 32 * kh + 8 * lhalf);
            const v8h bhi = *(const v8h*)(rb + 32 * kh + 16 + 8 * lhalf);
            rpa[kh] = __builtin_shufflevector(alo, ahi,
                0, 1, 2, 3, 4, 5, 6, 7, 8, 9, 10, 11, 12, 13, 14, 15);
            rpb[kh] = __builtin_shufflevector(blo, bhi,
                0, 1, 2, 3, 4, 5, 6, 7, 8, 9, 10, 11, 12, 13, 14, 15);
        }

        // A = relu(tp + rp), all packed f16
        const v16h A0a = __builtin_elementwise_max(tph[0] + rpa[0], zf);
        const v16h A1a = __builtin_elementwise_max(tph[1] + rpa[1], zf);
        const v16h A0b = __builtin_elementwise_max(tph[0] + rpb[0], zf);
        const v16h A1b = __builtin_elementwise_max(tph[1] + rpb[1], zf);

        v8f D0a = C0, D1a = C1, D0b = C0, D1b = C1;
        D0a = __builtin_amdgcn_wmma_f32_16x16x32_f16(false, A0a, false, Bop[0][0], (short)0, D0a, false, false);
        D1a = __builtin_amdgcn_wmma_f32_16x16x32_f16(false, A0a, false, Bop[0][1], (short)0, D1a, false, false);
        D0b = __builtin_amdgcn_wmma_f32_16x16x32_f16(false, A0b, false, Bop[0][0], (short)0, D0b, false, false);
        D1b = __builtin_amdgcn_wmma_f32_16x16x32_f16(false, A0b, false, Bop[0][1], (short)0, D1b, false, false);
        D0a = __builtin_amdgcn_wmma_f32_16x16x32_f16(false, A1a, false, Bop[1][0], (short)0, D0a, false, false);
        D1a = __builtin_amdgcn_wmma_f32_16x16x32_f16(false, A1a, false, Bop[1][1], (short)0, D1a, false, false);
        D0b = __builtin_amdgcn_wmma_f32_16x16x32_f16(false, A1b, false, Bop[1][0], (short)0, D0b, false, false);
        D1b = __builtin_amdgcn_wmma_f32_16x16x32_f16(false, A1b, false, Bop[1][1], (short)0, D1b, false, false);

        // per-lane score contributions: x[q], q = robot_sel*8 + task_row_p
        float x[16];
        #pragma unroll
        for (int p = 0; p < 8; ++p) {
            x[p]     = relu1(D0a[p]) * w3lo + relu1(D1a[p]) * w3hi;
            x[8 + p] = relu1(D0b[p]) * w3lo + relu1(D1b[p]) * w3hi;
        }

        // reduce-scatter butterfly over the 16-lane half: 8+4+2+1 permutes
        {
            const bool s8 = (lane & 8) != 0;
            #pragma unroll
            for (int i = 0; i < 8; ++i) {
                const float send = s8 ? x[i] : x[i + 8];
                const float keep = s8 ? x[i + 8] : x[i];
                x[i] = keep + __shfl_xor(send, 8, 32);
            }
            const bool s4 = (lane & 4) != 0;
            #pragma unroll
            for (int i = 0; i < 4; ++i) {
                const float send = s4 ? x[i] : x[i + 4];
                const float keep = s4 ? x[i + 4] : x[i];
                x[i] = keep + __shfl_xor(send, 4, 32);
            }
            const bool s2 = (lane & 2) != 0;
            #pragma unroll
            for (int i = 0; i < 2; ++i) {
                const float send = s2 ? x[i] : x[i + 2];
                const float keep = s2 ? x[i + 2] : x[i];
                x[i] = keep + __shfl_xor(send, 2, 32);
            }
            const bool s1 = (lane & 1) != 0;
            {
                const float send = s1 ? x[0] : x[1];
                const float keep = s1 ? x[1] : x[0];
                x[0] = keep + __shfl_xor(send, 1, 32);
            }
        }

        // lane l holds score for task row m = 8*lhalf + (l&7), robot r0 + ((l>>3)&1)
        const int m  = lhalf * 8 + (lane & 7);
        const int rr = r0 + ((lane >> 3) & 1);
        sc[m][rr] = x[0];
    }
    __syncthreads();

    // softmax over robots per task row: 16 threads per row, 32 elems each
    const int row = tid >> 4;
    const int seg = tid & 15;
    float* srow = sc[row];
    float mx = -3.0e38f;
    for (int i = 0; i < 32; ++i) mx = fmaxf(mx, srow[seg * 32 + i]);
    #pragma unroll
    for (int msk = 1; msk < 16; msk <<= 1) mx = fmaxf(mx, __shfl_xor(mx, msk, 32));
    float sum = 0.f;
    for (int i = 0; i < 32; ++i) {
        const float e = __expf(srow[seg * 32 + i] - mx);
        srow[seg * 32 + i] = e;
        sum += e;
    }
    #pragma unroll
    for (int msk = 1; msk < 16; msk <<= 1) sum += __shfl_xor(sum, msk, 32);
    const float inv = 1.f / sum;
    float* orow = out + ((long)(b * T_) + t0 + row) * R_ + seg * 32;
    for (int i = 0; i < 32; i += 4) {
        float4 v;
        v.x = srow[seg * 32 + i + 0] * inv;
        v.y = srow[seg * 32 + i + 1] * inv;
        v.z = srow[seg * 32 + i + 2] * inv;
        v.w = srow[seg * 32 + i + 3] * inv;
        *(float4*)(orow + i) = v;
    }
}

// ---------------------------------------------------------------------------
extern "C" void kernel_launch(void* const* d_in, const int* in_sizes, int n_in,
                              void* d_out, int out_size, void* d_ws, size_t ws_size,
                              hipStream_t stream) {
    (void)in_sizes; (void)n_in; (void)out_size; (void)ws_size;
    const float* robot_states = (const float*)d_in[0];
    const float* task_states  = (const float*)d_in[1];
    const float* r_w1 = (const float*)d_in[2];  const float* r_b1 = (const float*)d_in[3];
    const float* r_w2 = (const float*)d_in[4];  const float* r_b2 = (const float*)d_in[5];
    const float* t_w1 = (const float*)d_in[6];  const float* t_b1 = (const float*)d_in[7];
    const float* t_w2 = (const float*)d_in[8];  const float* t_b2 = (const float*)d_in[9];
    const float* wq   = (const float*)d_in[10]; const float* bq   = (const float*)d_in[11];
    const float* wk   = (const float*)d_in[12]; const float* bk   = (const float*)d_in[13];
    const float* wv   = (const float*)d_in[14]; const float* bv   = (const float*)d_in[15];
    const float* wo   = (const float*)d_in[16]; const float* bo   = (const float*)d_in[17];
    const float* a_w1 = (const float*)d_in[18]; const float* a_b1 = (const float*)d_in[19];
    const float* a_w2 = (const float*)d_in[20]; const float* a_b2 = (const float*)d_in[21];
    const float* a_w3 = (const float*)d_in[22];
    // d_in[23] = a_b3: constant shift before softmax -> cancels, unused.
    // d_in[24], d_in[25]: num_robots=512, num_tasks=128 (full slices, hardcoded)

    // workspace layout: f32 buffers then f16 buffers (~13 MB total)
    float* ws   = (float*)d_ws;
    float* kmat = ws;                           // B*R*H = 1048576 f32
    float* vmat = kmat + B_ * R_ * H_;          // 1048576 f32
    float* qbuf = vmat + B_ * R_ * H_;          // B*T*H = 262144 f32
    float* ctx  = qbuf + B_ * T_ * H_;          // 262144 f32
    _Float16* rp_h = (_Float16*)(ctx + B_ * T_ * H_);   // B*R*H halves
    _Float16* tp_h = rp_h + B_ * R_ * H_;               // B*T*H halves

    robot_encoder_kernel<<<B_ * (R_ / 64), 256, 0, stream>>>(
        robot_states, r_w1, r_b1, r_w2, r_b2, wk, bk, wv, bv, a_w1, kmat, vmat, rp_h);
    task_encoder_kernel<<<B_ * (T_ / 64), 256, 0, stream>>>(
        task_states, t_w1, t_b1, t_w2, t_b2, wq, bq, qbuf);
    attn_kernel<<<B_ * 4, 128, 0, stream>>>(qbuf, kmat, vmat, ctx);
    outproj_tp_kernel<<<B_ * (T_ / 64), 256, 0, stream>>>(ctx, wo, bo, a_w1, a_b1, tp_h);
    pairwise_kernel<<<B_ * (T_ / 16), 256, 0, stream>>>(
        tp_h, rp_h, a_w2, a_b2, a_w3, (float*)d_out);
}
